// DenseAqt_36953898614819
// MI455X (gfx1250) — 8-bit matrix operand order repaired and checked by reading, compiled, not run on hardware
//
#include <hip/hip_runtime.h>

#define BB 131072
#define DD 512
#define FF 512
#define KBLKS (DD / 64)

typedef int  v8i __attribute__((ext_vector_type(8)));
typedef int  v4i __attribute__((ext_vector_type(4)));


__global__ __launch_bounds__(256) void quant_w_kernel(
    const float* __restrict__ w, signed char* __restrict__ wq_sh,
    float* __restrict__ deq) {
  const int f = blockIdx.x;
  const int t = threadIdx.x;

  const float v0 = w[(size_t)t * FF + f];
  const float v1 = w[(size_t)(t + 256) * FF + f];

  __shared__ float red[256];
  red[t] = fmaxf(fabsf(v0), fabsf(v1));
  __syncthreads();
#pragma unroll
  for (int s = 128; s > 0; s >>= 1) {
    if (t < s) red[t] = fmaxf(red[t], red[t + s]);
    __syncthreads();
  }
  const float wb = fmaxf(red[0], 1e-6f);
  const float ws = 127.0f / wb;

  const int n_blk = f >> 4;
  const int n     = f & 15;
#pragma unroll
  for (int e = 0; e < 2; ++e) {
    const int d = t + 256 * e;
    float r = floorf((e ? v1 : v0) * ws + 0.5f);
    r = fminf(fmaxf(r, -127.0f), 127.0f);
    const int kb   = d >> 6;
    const int kin  = d & 63;
    const int half = (kin >> 3) & 1;
    const int j    = (kin >> 4) & 3;
    const int b    = kin & 7;
    wq_sh[(((size_t)n_blk * KBLKS + kb) * 32 + (n + 16 * half)) * 32 + j * 8 + b] =
        (signed char)(int)r;
  }
  if (t == 0) deq[f] = (6.0f * wb) / (127.0f * 127.0f);
}

__device__ __forceinline__ unsigned int quant4(const float4 v, float s) {
  float a = fminf(fmaxf(floorf(v.x * s + 0.5f), -127.0f), 127.0f);
  float b = fminf(fmaxf(floorf(v.y * s + 0.5f), -127.0f), 127.0f);
  float c = fminf(fmaxf(floorf(v.z * s + 0.5f), -127.0f), 127.0f);
  float d = fminf(fmaxf(floorf(v.w * s + 0.5f), -127.0f), 127.0f);
  int ia = (int)a, ib = (int)b, ic = (int)c, id = (int)d;
  return (unsigned int)((ia & 255) | ((ib & 255) << 8) | ((ic & 255) << 16) |
                        ((id & 255) << 24));
}

__global__ __launch_bounds__(256) void quant_x_kernel(
    const float* __restrict__ x, signed char* __restrict__ xq_sh) {
  __shared__ unsigned long long sh[1024];
  const float a_scale = 127.0f / 6.0f;
  const int t = threadIdx.x;
  const float* xb = x + (size_t)blockIdx.x * 16 * DD;

#pragma unroll
  for (int r = 0; r < 4; ++r) {
    const int j = t + 256 * r;
    const int m = j >> 6;
    const int c = j & 63;
    const float4* p = (const float4*)(xb + m * DD + c * 8);
    float4 lo = p[0];
    float4 hi = p[1];
    unsigned long long packed =
        (unsigned long long)quant4(lo, a_scale) |
        ((unsigned long long)quant4(hi, a_scale) << 32);
    const int off8 =
        ((c >> 3) * 1024 + (m + 16 * (c & 1)) * 32 + ((c >> 1) & 3) * 8) >> 3;
    sh[off8] = packed;
  }
  __syncthreads();

  const v4i* s = (const v4i*)sh;
  v4i* dst = (v4i*)(xq_sh + (size_t)blockIdx.x * 8192);
  dst[t * 2 + 0] = s[t * 2 + 0];
  dst[t * 2 + 1] = s[t * 2 + 1];
}

__global__ __launch_bounds__(256) void gemm_iu8_kernel(
    const signed char* __restrict__ xq_sh, const signed char* __restrict__ wq_sh,
    const float* __restrict__ deq, const float* __restrict__ bias,
    float* __restrict__ out) {
  const int lane = threadIdx.x & 31;
  const int wave = threadIdx.x >> 5;
  const int wm = wave & 1;
  const int wn = wave >> 1;

  const int m0 = blockIdx.y * 128 + wm * 64;
  const int n0 = blockIdx.x * 128 + wn * 32;
  const int tm0 = m0 >> 4;
  const int tn0 = n0 >> 4;

  const int ln   = lane & 15;
  const int half = lane >> 4;

  v8i acc[4][2];
#pragma unroll
  for (int mi = 0; mi < 4; ++mi)
#pragma unroll
    for (int ni = 0; ni < 2; ++ni)
#pragma unroll
      for (int r = 0; r < 8; ++r) acc[mi][ni][r] = 0;

  const signed char* abase[4];
#pragma unroll
  for (int mi = 0; mi < 4; ++mi)
    abase[mi] = xq_sh + ((size_t)(tm0 + mi) * KBLKS) * 1024 + lane * 32;
  const signed char* bbase[2];
#pragma unroll
  for (int ni = 0; ni < 2; ++ni)
    bbase[ni] = wq_sh + ((size_t)(tn0 + ni) * KBLKS) * 1024 + lane * 32;

#pragma unroll
  for (int kb = 0; kb < KBLKS; ++kb) {
    __builtin_prefetch(abase[0] + (kb + 1) * 1024, 0, 1);

    v8i a[4];
#pragma unroll
    for (int mi = 0; mi < 4; ++mi) {
      union { v8i v; v4i h[2]; } ua;
      const signed char* p = abase[mi] + kb * 1024;
      ua.h[0] = *(const v4i*)(p + 0);
      ua.h[1] = *(const v4i*)(p + 16);
      a[mi] = ua.v;
    }

    v8i b[2];
#pragma unroll
    for (int ni = 0; ni < 2; ++ni) {
      union { v8i v; v4i h[2]; } ub;
      const signed char* p = bbase[ni] + kb * 1024;
      ub.h[0] = *(const v4i*)(p + 0);
      ub.h[1] = *(const v4i*)(p + 16);
      b[ni] = ub.v;
    }

#pragma unroll
    for (int mi = 0; mi < 4; ++mi)
#pragma unroll
      for (int ni = 0; ni < 2; ++ni)
        acc[mi][ni] = __builtin_amdgcn_wmma_i32_16x16x64_iu8(
             true, a[mi],  true, b[ni], acc[mi][ni],
             false,  false);
  }

#pragma unroll
  for (int ni = 0; ni < 2; ++ni) {
    const int col = n0 + ni * 16 + ln;
    const float dq = deq[col];
    const float bs = bias[col];
#pragma unroll
    for (int mi = 0; mi < 4; ++mi) {
      const int mbase = m0 + mi * 16 + half * 8;
#pragma unroll
      for (int r = 0; r < 8; ++r) {
        out[(size_t)(mbase + r) * FF + col] = (float)acc[mi][ni][r] * dq + bs;
      }
    }
  }
}

extern "C" void kernel_launch(void* const* d_in, const int* in_sizes, int n_in,
                              void* d_out, int out_size, void* d_ws, size_t ws_size,
                              hipStream_t stream) {
  const float* x      = (const float*)d_in[0];
  const float* kernel = (const float*)d_in[1];
  const float* bias   = (const float*)d_in[2];

  float* out = (float*)d_out;

  char* ws = (char*)d_ws;
  signed char* xq_sh = (signed char*)ws;
  signed char* wq_sh = (signed char*)(ws + (size_t)BB * DD);
  float*       deq   = (float*)(ws + (size_t)BB * DD + (size_t)FF * DD);

  quant_w_kernel<<<FF, 256, 0, stream>>>(kernel, wq_sh, deq);
  quant_x_kernel<<<BB / 16, 256, 0, stream>>>(x, xq_sh);

  dim3 grid(FF / 128, BB / 128);
  gemm_iu8_kernel<<<grid, 256, 0, stream>>>(xq_sh, wq_sh, deq, bias, out);
}
